// CustomLSTMLayer_12618613916356
// MI455X (gfx1250) — matrix operand order repaired and checked by reading, not compiled, not run on hardware
//
#include <hip/hip_runtime.h>

// ---------------------------------------------------------------------------
// Fused LSTM cell for MI455X (gfx1250, wave32, WMMA + Tensor Data Mover).
//   gates = [x | h_prev] (bf16) @ [W ; V]^T (bf16)  -> f32 accum (WMMA)
//   epilogue (in-register): c_next = sig(f)*c_prev + sig(i)*tanh(c~)
//                           h_next = sig(o)*tanh(c_next)
// Staging: TDM tensor_load_to_lds (double buffered), TENSORcnt-synchronized.
// ---------------------------------------------------------------------------

#define B_SZ 4096
#define I_SZ 2048
#define H_SZ 2048
#define K_SZ (I_SZ + H_SZ)   // 4096 (concatenated K)
#define NG   (4 * H_SZ)      // 8192 (4 gates stacked in N)

// GEMM tiling
#define BM  128              // block rows (batch)
#define BN  32               // block cols per gate (hidden)
#define KT  64               // K-tile staged in LDS
#define LDA 72               // padded LDS stride (elements): 64 data + 8 pad
#define LDB 72

typedef __attribute__((ext_vector_type(16))) __bf16 v16bf;
typedef __attribute__((ext_vector_type(8)))  float  v8f;
typedef __attribute__((ext_vector_type(4)))  unsigned int u32x4;
typedef __attribute__((ext_vector_type(8)))  int          i32x8;
typedef __attribute__((ext_vector_type(4)))  int          i32x4;

union Frag { v16bf v; uint4 q[2]; };

static __device__ __forceinline__ unsigned short f2bf(float f) {
    union { float f; unsigned int u; } v; v.f = f;
    unsigned int r = v.u + 0x7FFFu + ((v.u >> 16) & 1u);   // round-to-nearest-even
    return (unsigned short)(r >> 16);
}

static __device__ __forceinline__ float fsigmoid(float x) {
    return 1.0f / (1.0f + __expf(-x));
}

#if defined(__has_builtin)
#if __has_builtin(__builtin_amdgcn_tensor_load_to_lds) && \
    __has_builtin(__builtin_amdgcn_s_wait_tensorcnt)
#define USE_TDM 1
#endif
#endif
#ifndef USE_TDM
#define USE_TDM 0
#endif

#if USE_TDM
// ---------------------------------------------------------------------------
// TDM descriptor build + issue (CDNA5 ISA ch.8, D# groups 0..3).
// Loads a (tile_d0 x tile_d1 [x tile_d2]) tile of 2-byte elements into LDS,
// inserting 4 DWORDs of padding after every 32 DWORDs (=> 72-element rows).
// clang-23 (amdgpu-toolchain) builtin arity: 6 args
//   (u32x4 g0, i32x8 g1, i32x4 g2, i32x4 g3, i32x8 ext, i32 cpol)
// ---------------------------------------------------------------------------
static __device__ __forceinline__ void tdm_load_tile(
    unsigned lds_byte_addr, const void* gptr,
    unsigned tensor_d0, unsigned tensor_d1, unsigned tensor_d2,
    unsigned tile_d0, unsigned tile_d1, unsigned tile_d2,
    unsigned long long d0_stride, unsigned long long d1_stride)
{
    unsigned long long ga = (unsigned long long)gptr;
    u32x4 g0 = {
        1u,                                                   // count=1 (valid, user)
        lds_byte_addr,                                        // lds_addr[31:0]
        (unsigned)(ga & 0xFFFFFFFFu),                         // global_addr[31:0]
        (unsigned)((ga >> 32) & 0x1FFFFFFu) | (2u << 30)      // addr[56:32] | type=2
    };
    i32x8 g1 = {
        (int)((1u << 16)      // data_size = 2 bytes
            | (1u << 20)      // pad_enable
            | (4u << 22)      // pad_interval: 2^(4+1)=32 DWORDs (=64 bf16)
            | (3u << 25)),    // pad_amount: 3+1 = 4 DWORDs (=8 bf16)
        (int)((tensor_d0 & 0xFFFFu) << 16),                           // dim0[15:0]
        (int)((tensor_d0 >> 16) | ((tensor_d1 & 0xFFFFu) << 16)),     // dim0[31:16]|dim1[15:0]
        (int)((tensor_d1 >> 16) | (tile_d0 << 16)),                   // dim1[31:16]|tile0
        (int)(tile_d1 | (tile_d2 << 16)),                             // tile1|tile2
        (int)(unsigned)(d0_stride & 0xFFFFFFFFu),                     // d0_stride[31:0]
        (int)(((unsigned)(d0_stride >> 32) & 0xFFFFu)
            | (((unsigned)d1_stride & 0xFFFFu) << 16)),               // d0s[47:32]|d1s[15:0]
        (int)(unsigned)((d1_stride >> 16) & 0xFFFFFFFFu)              // d1_stride[47:16]
    };
    i32x4 g2 = { (int)tensor_d2, 0, 0, 0 };   // tensor_dim2; dim3/iterate unused
    i32x4 g3 = { 0, 0, 0, 0 };
    i32x8 gx = { 0, 0, 0, 0, 0, 0, 0, 0 };    // extension group (unused, zeros)
    __builtin_amdgcn_tensor_load_to_lds(g0, g1, g2, g3, gx, 0);
}

static __device__ __forceinline__ void tdm_issue_tile(
    const unsigned short* U, const unsigned short* WV,
    unsigned ldsA, unsigned ldsB, int bm0, int n0, int k0)
{
    // A tile: 2D, 64(k) x 128(batch rows) from U[bm0.., k0..]
    tdm_load_tile(ldsA, U + (size_t)bm0 * K_SZ + k0,
                  K_SZ, B_SZ, 1, KT, BM, 0, K_SZ, 0ull);
    // B tile: 3D, 64(k) x 32(hidden rows) x 4(gates) from WV[g*H+n0.., k0..]
    tdm_load_tile(ldsB, WV + (size_t)n0 * K_SZ + k0,
                  K_SZ, H_SZ, 4, KT, BN, 4,
                  K_SZ, (unsigned long long)H_SZ * K_SZ);
}
#endif // USE_TDM

// ---------------------------------------------------------------------------
// Pack kernels: fp32 -> bf16, concatenating along K.
// ---------------------------------------------------------------------------
__global__ void pack_U(const float* __restrict__ x, const float* __restrict__ h,
                       unsigned short* __restrict__ U) {
    int idx = blockIdx.x * blockDim.x + threadIdx.x;   // over B_SZ*K_SZ
    if (idx >= B_SZ * K_SZ) return;
    int b = idx >> 12;               // / 4096
    int k = idx & (K_SZ - 1);
    float v = (k < I_SZ) ? x[b * I_SZ + k] : h[b * H_SZ + (k - I_SZ)];
    U[idx] = f2bf(v);
}

__global__ void pack_WV(const float* __restrict__ W, const float* __restrict__ V,
                        unsigned short* __restrict__ WV) {
    long long idx = (long long)blockIdx.x * blockDim.x + threadIdx.x; // NG*K_SZ
    if (idx >= (long long)NG * K_SZ) return;
    int n = (int)(idx >> 12);
    int k = (int)(idx & (K_SZ - 1));
    float v = (k < I_SZ) ? W[(size_t)n * I_SZ + k]
                         : V[(size_t)n * H_SZ + (k - I_SZ)];
    WV[idx] = f2bf(v);
}

// ---------------------------------------------------------------------------
// Fused GEMM + LSTM epilogue. Grid: (H/BN, B/BM). 256 threads = 8 waves.
// Double-buffered LDS; wave 0 drives the TDM, all waves run WMMA.
// ---------------------------------------------------------------------------
__global__ __launch_bounds__(256, 1)
void lstm_wmma(const unsigned short* __restrict__ U,
               const unsigned short* __restrict__ WV,
               const float* __restrict__ c_prev,
               const float* __restrict__ bW, const float* __restrict__ bV,
               const float* __restrict__ bb,
               float* __restrict__ out /* [h_next | c_next] */) {
    __shared__ unsigned short As[2][BM * LDA];        // 2 x 128x64 bf16 (+pad)
    __shared__ unsigned short Bs[2][4][BN * LDB];     // 2 x 4 gates x 32x64 bf16

    const int tid  = threadIdx.x;
    const int lane = tid & 31;
    const int wid  = tid >> 5;
    const int wm   = wid & 3;        // wave's M group (32 rows each)
    const int wn   = wid >> 2;       // wave's N group (16 cols each)
    const int bm0  = blockIdx.y * BM;
    const int n0   = blockIdx.x * BN;

    v8f acc[4][2];
    #pragma unroll
    for (int g = 0; g < 4; ++g)
        #pragma unroll
        for (int mt = 0; mt < 2; ++mt)
            #pragma unroll
            for (int r = 0; r < 8; ++r) acc[g][mt][r] = 0.0f;

    const int hiA = (lane >> 4) << 3;    // +8 elements for lanes 16..31 (A frag)
    const int hiB = (lane >> 4) << 3;    // +8 elements for lanes 16..31 (B frag)
    const int l15 = lane & 15;
    const int NT  = K_SZ / KT;           // 64 K-tiles

#if USE_TDM
    // scalar branch: only wave 0 issues TDM ops (TDM ignores EXEC)
    const bool issuer = (__builtin_amdgcn_readfirstlane(tid) == 0);
    const unsigned ldsA[2] = { (unsigned)(size_t)&As[0][0],
                               (unsigned)(size_t)&As[1][0] };
    const unsigned ldsB[2] = { (unsigned)(size_t)&Bs[0][0][0],
                               (unsigned)(size_t)&Bs[1][0][0] };
    if (issuer) tdm_issue_tile(U, WV, ldsA[0], ldsB[0], bm0, n0, 0);
#endif

    for (int t = 0; t < NT; ++t) {
        const int cur = t & 1;
#if USE_TDM
        if (issuer) {
            if (t + 1 < NT) {
                tdm_issue_tile(U, WV, ldsA[cur ^ 1], ldsB[cur ^ 1],
                               bm0, n0, (t + 1) * KT);
                __builtin_amdgcn_s_wait_tensorcnt(2);  // tile t done (in-order)
            } else {
                __builtin_amdgcn_s_wait_tensorcnt(0);
            }
        }
#else
        {   // fallback: synchronous staging through VGPRs
            const int k0 = t * KT;
            #pragma unroll
            for (int u = 0; u < 4; ++u) {
                int c   = tid + u * 256;
                int row = c >> 3;
                int col = (c & 7) * 8;
                uint4 d = *(const uint4*)(U + (size_t)(bm0 + row) * K_SZ + k0 + col);
                *(uint4*)(As[cur] + row * LDA + col) = d;
            }
            #pragma unroll
            for (int u = 0; u < 4; ++u) {
                int c   = tid + u * 256;
                int g   = c >> 8;
                int r   = (c >> 3) & 31;
                int col = (c & 7) * 8;
                uint4 d = *(const uint4*)(WV + (size_t)(g * H_SZ + n0 + r) * K_SZ + k0 + col);
                *(uint4*)(&Bs[cur][g][r * LDB + col]) = d;
            }
        }
#endif
        __syncthreads();

        const unsigned short* Ab = As[cur];
        #pragma unroll
        for (int s = 0; s < 2; ++s) {        // two K=32 WMMA steps per tile
            const int ks = s * 32;
            // A frag (16x32 bf16): lane<16 -> K{0-7,16-23}, lane>=16 -> K{8-15,24-31}
            Frag a[2];
            #pragma unroll
            for (int mt = 0; mt < 2; ++mt) {
                const unsigned short* p =
                    Ab + (wm * 32 + mt * 16 + l15) * LDA + ks + hiA;
                a[mt].q[0] = *(const uint4*)p;
                a[mt].q[1] = *(const uint4*)(p + 16);
            }
            #pragma unroll
            for (int g = 0; g < 4; ++g) {
                // B frag (32x16 bf16): lane = column;
                // lo lanes = K{0-7,16-23}, hi lanes = K{8-15,24-31}
                const unsigned short* p =
                    &Bs[cur][g][(wn * 16 + l15) * LDB + ks + hiB];
                Frag bf;
                bf.q[0] = *(const uint4*)p;
                bf.q[1] = *(const uint4*)(p + 16);
                #pragma unroll
                for (int mt = 0; mt < 2; ++mt) {
                    acc[g][mt] = __builtin_amdgcn_wmma_f32_16x16x32_bf16(
                        false, a[mt].v, false, bf.v,
                        (short)0, acc[g][mt], false, false);
                }
            }
        }
        __syncthreads();
    }

    // ---- fused LSTM epilogue (C/D layout: VGPR r -> row r (+8 hi lanes),
    //      lane%16 -> column) ----
    const int col = n0 + wn * 16 + l15;
    float bias[4];
    #pragma unroll
    for (int g = 0; g < 4; ++g) {
        int gi = g * H_SZ + col;
        bias[g] = bW[gi] + bV[gi] + bb[gi];
    }
    const int rbase = bm0 + wm * 32 + ((lane >> 4) << 3);
    #pragma unroll
    for (int mt = 0; mt < 2; ++mt) {
        #pragma unroll
        for (int r = 0; r < 8; ++r) {
            int row = rbase + mt * 16 + r;
            float f_t = fsigmoid(acc[0][mt][r] + bias[0]);
            float i_t = fsigmoid(acc[1][mt][r] + bias[1]);
            float o_t = fsigmoid(acc[2][mt][r] + bias[2]);
            float cg  = tanhf(acc[3][mt][r] + bias[3]);
            float cp  = c_prev[(size_t)row * H_SZ + col];
            float cn  = f_t * cp + i_t * cg;
            float hn  = o_t * tanhf(cn);
            out[(size_t)row * H_SZ + col] = hn;                              // h_next
            out[(size_t)B_SZ * H_SZ + (size_t)row * H_SZ + col] = cn;        // c_next
        }
    }
}

// ---------------------------------------------------------------------------
extern "C" void kernel_launch(void* const* d_in, const int* in_sizes, int n_in,
                              void* d_out, int out_size, void* d_ws, size_t ws_size,
                              hipStream_t stream) {
    const float* x   = (const float*)d_in[0];
    const float* h   = (const float*)d_in[1];
    const float* cp  = (const float*)d_in[2];
    const float* W   = (const float*)d_in[3];
    const float* bW  = (const float*)d_in[4];
    const float* V   = (const float*)d_in[5];
    const float* bV  = (const float*)d_in[6];
    const float* bb  = (const float*)d_in[7];
    float* out = (float*)d_out;

    unsigned short* U  = (unsigned short*)d_ws;                                   // 32 MB
    unsigned short* WV = (unsigned short*)((char*)d_ws + (size_t)B_SZ * K_SZ * 2);// 64 MB

    {
        int total = B_SZ * K_SZ;
        pack_U<<<(total + 255) / 256, 256, 0, stream>>>(x, h, U);
    }
    {
        long long total = (long long)NG * K_SZ;
        pack_WV<<<(unsigned)((total + 255) / 256), 256, 0, stream>>>(W, V, WV);
    }
    dim3 grid(H_SZ / BN, B_SZ / BM);   // 64 x 32 blocks
    lstm_wmma<<<grid, 256, 0, stream>>>(U, WV, cp, bW, bV, bb, out);
}
